// DotProductNonLocalBlock_36103495090207
// MI455X (gfx1250) — matrix operand order repaired and checked by reading, not compiled, not run on hardware
//
#include <hip/hip_runtime.h>
#include <hip/hip_bf16.h>

// ---------------------------------------------------------------------------
// DotProductNonLocalBlock for MI455X (gfx1250, wave32, WMMA).
// Reassociated linear attention: embed = Q * (K^T V) / N  (no softmax in ref).
// All GEMMs run on v_wmma_f32_16x16x32_bf16 with 2x2 register blocking per
// wave (4 independent accumulators -> hides XDL latency, halves operand
// traffic per FLOP).
// ---------------------------------------------------------------------------

typedef __attribute__((ext_vector_type(16))) __bf16 v16bf;
typedef __attribute__((ext_vector_type(8)))  __bf16 v8bf;
typedef __attribute__((ext_vector_type(8)))  float  v8f;
typedef __attribute__((ext_vector_type(4)))  float  v4f;

constexpr int B_ = 8;
constexpr int C_ = 512;
constexpr int N_ = 3136;   // 56*56 = 98 blocks of 32
constexpr int E_ = 256;

// workspace layout (bytes)
constexpr size_t OFF_XT  = 0;                                        // [B][N][C] bf16
constexpr size_t OFF_Q   = OFF_XT  + (size_t)B_ * N_ * C_ * 2;       // [B][N][E] bf16
constexpr size_t OFF_K   = OFF_Q   + (size_t)B_ * N_ * E_ * 2;       // [B][E][N] bf16
constexpr size_t OFF_V   = OFF_K   + (size_t)B_ * E_ * N_ * 2;       // [B][E][N] bf16
constexpr size_t OFF_MT  = OFF_V   + (size_t)B_ * E_ * N_ * 2;       // [B][E][E] bf16 (transposed M)
constexpr size_t OFF_EMB = OFF_MT  + (size_t)B_ * E_ * E_ * 2;       // [B][N][E] bf16
constexpr size_t OFF_W   = OFF_EMB + (size_t)B_ * N_ * E_ * 2;       // wq|wk|wv|wo bf16, each E*C
// total ~79.2 MB

__device__ __forceinline__ __bf16 f2bf(float f) {
  union { float f; unsigned u; } c; c.f = f;
  unsigned u = c.u + 0x7FFFu + ((c.u >> 16) & 1u);   // round-to-nearest-even
  union { unsigned short s; __bf16 b; } o; o.s = (unsigned short)(u >> 16);
  return o.b;
}

// A fragment: 16x32 (MxK), element (m,k) at base + m*ld + k.
// ISA layout: lane m = l&15; lanes<16 hold K 0..7 & 16..23, lanes>=16 hold 8..15 & 24..31.
__device__ __forceinline__ v16bf load_a(const __bf16* base, int ld, int row0,
                                        int k0, int lane) {
  const __bf16* p = base + (size_t)(row0 + (lane & 15)) * ld + k0 + ((lane >> 4) & 1) * 8;
  v8bf lo = *(const v8bf*)(p);
  v8bf hi = *(const v8bf*)(p + 16);
  return __builtin_shufflevector(lo, hi, 0,1,2,3,4,5,6,7,8,9,10,11,12,13,14,15);
}

// B fragment: 32x16 (KxN). Value B[k][col] lives at base + col*ld + k (col-major B).
// ISA layout: lane col = l&15; lanes<16 hold K 0..7 & 16..23, lanes>=16 hold 8..15 & 24..31.
__device__ __forceinline__ v16bf load_b(const __bf16* base, int ld, int col0,
                                        int k0, int lane) {
  const __bf16* p = base + (size_t)(col0 + (lane & 15)) * ld + k0 + ((lane >> 4) & 1) * 8;
  v8bf lo = *(const v8bf*)(p);
  v8bf hi = *(const v8bf*)(p + 16);
  return __builtin_shufflevector(lo, hi, 0,1,2,3,4,5,6,7,8,9,10,11,12,13,14,15);
}

struct Acc4 { v8f d00, d01, d10, d11; };   // d[rowtile][coltile]

// 32x32 output block per wave: 2 A frags x 2 B frags -> 4 independent WMMAs.
__device__ __forceinline__ Acc4 gemm_tile_2x2(const __bf16* A, int lda, int row0,
                                              const __bf16* Bm, int ldb, int col0,
                                              int K, int lane) {
  Acc4 t;
  t.d00 = (v8f){0.f,0.f,0.f,0.f,0.f,0.f,0.f,0.f};
  t.d01 = t.d00; t.d10 = t.d00; t.d11 = t.d00;
  #pragma unroll 2
  for (int k0 = 0; k0 < K; k0 += 32) {
    v16bf a0 = load_a(A, lda, row0,      k0, lane);
    v16bf a1 = load_a(A, lda, row0 + 16, k0, lane);
    v16bf b0 = load_b(Bm, ldb, col0,      k0, lane);
    v16bf b1 = load_b(Bm, ldb, col0 + 16, k0, lane);
    t.d00 = __builtin_amdgcn_wmma_f32_16x16x32_bf16(false, a0, false, b0, (short)0, t.d00, false, false);
    t.d01 = __builtin_amdgcn_wmma_f32_16x16x32_bf16(false, a0, false, b1, (short)0, t.d01, false, false);
    t.d10 = __builtin_amdgcn_wmma_f32_16x16x32_bf16(false, a1, false, b0, (short)0, t.d10, false, false);
    t.d11 = __builtin_amdgcn_wmma_f32_16x16x32_bf16(false, a1, false, b1, (short)0, t.d11, false, false);
  }
  return t;
}

// Packed bf16 store of one 16x16 tile: dst[(col)*ld + row], rows contiguous.
__device__ __forceinline__ void store_bf8(__bf16* dst, int ld, int row0, int col0,
                                          int lane, const v8f acc, float scale, float add) {
  int hi8 = ((lane >> 4) & 1) * 8, l15 = lane & 15;
  v8bf s;
  #pragma unroll
  for (int i = 0; i < 8; ++i) s[i] = f2bf(acc[i] * scale + add);
  *(v8bf*)(dst + (size_t)(col0 + l15) * ld + row0 + hi8) = s;
}

__device__ __forceinline__ void store_bf8_rowbias(__bf16* dst, int ld, int row0, int col0,
                                                  int lane, const v8f acc,
                                                  const float* __restrict__ bias) {
  int hi8 = ((lane >> 4) & 1) * 8, l15 = lane & 15;
  const float* bp = bias + row0 + hi8;
  v8bf s;
  #pragma unroll
  for (int i = 0; i < 8; ++i) s[i] = f2bf(acc[i] + bp[i]);
  *(v8bf*)(dst + (size_t)(col0 + l15) * ld + row0 + hi8) = s;
}

// ---- stage 1: x[b][c][n] f32 -> xt[b][n][c] bf16 (LDS tiled transpose) -----
__global__ void cvt_x_kernel(const float* __restrict__ x, __bf16* __restrict__ xt) {
  __shared__ float tile[32][33];
  int b  = blockIdx.z;
  int c0 = blockIdx.y * 32;
  int n0 = blockIdx.x * 32;
  const float* xb = x + (size_t)b * C_ * N_;
  #pragma unroll
  for (int i = threadIdx.y; i < 32; i += 8)
    tile[i][threadIdx.x] = xb[(size_t)(c0 + i) * N_ + n0 + threadIdx.x];
  __syncthreads();
  __bf16* xtb = xt + (size_t)b * N_ * C_;
  #pragma unroll
  for (int i = threadIdx.y; i < 32; i += 8)
    xtb[(size_t)(n0 + i) * C_ + c0 + threadIdx.x] = f2bf(tile[threadIdx.x][i]);
}

// ---- stage 2: weights f32 -> bf16 [wq|wk|wv|wo] ---------------------------
__global__ void cvt_w_kernel(const float* __restrict__ wq, const float* __restrict__ wk,
                             const float* __restrict__ wv, const float* __restrict__ wo,
                             __bf16* __restrict__ dst) {
  int i = blockIdx.x * 256 + threadIdx.x;          // total 4 * E*C = 524288
  int seg = i >> 17;
  int off = i & (E_ * C_ - 1);
  const float* s = (seg == 0) ? wq : (seg == 1) ? wk : (seg == 2) ? wv : wo;
  dst[i] = f2bf(s[off]);
}

// ---- stage 3: Q/K/V projections (2x2 blocked) -----------------------------
// q[b][n][e] = sum_c wq[e][c] xt[b][n][c] + bq[e]   (rows=e, cols=n)
// k/v[b][e][n] = ...                                 (rows=n, cols=e)
__global__ void proj_kernel(const __bf16* __restrict__ xt, const __bf16* __restrict__ wbf,
                            const float* __restrict__ bq, const float* __restrict__ bk,
                            const float* __restrict__ bv,
                            __bf16* __restrict__ q, __bf16* __restrict__ k,
                            __bf16* __restrict__ v) {
  int wid  = blockIdx.x * 8 + (threadIdx.x >> 5);
  int lane = threadIdx.x & 31;
  int b     = wid / 2352;            // 3 * 8 * 98 blocks per batch
  int r     = wid % 2352;
  int which = r / 784;
  int tt    = r % 784;
  const __bf16* xb = xt + (size_t)b * N_ * C_;
  if (which == 0) {                                   // Q: rows=e, cols=n
    int et2 = tt / 98, nt2 = tt % 98;
    int row0 = et2 * 32, col0 = nt2 * 32;
    Acc4 t = gemm_tile_2x2(wbf, C_, row0, xb, C_, col0, C_, lane);
    __bf16* qb = q + (size_t)b * N_ * E_;
    store_bf8_rowbias(qb, E_, row0,      col0,      lane, t.d00, bq);
    store_bf8_rowbias(qb, E_, row0,      col0 + 16, lane, t.d01, bq);
    store_bf8_rowbias(qb, E_, row0 + 16, col0,      lane, t.d10, bq);
    store_bf8_rowbias(qb, E_, row0 + 16, col0 + 16, lane, t.d11, bq);
  } else {                                            // K/V: rows=n, cols=e
    const __bf16* wptr = wbf + (size_t)which * E_ * C_;
    const float*  bias = (which == 1) ? bk : bv;
    __bf16*       dstb = ((which == 1) ? k : v) + (size_t)b * E_ * N_;
    int nt2 = tt / 8, et2 = tt % 8;
    int row0 = nt2 * 32, col0 = et2 * 32;
    Acc4 t = gemm_tile_2x2(xb, C_, row0, wptr, C_, col0, C_, lane);
    int l15 = lane & 15;
    float bc0 = bias[col0 + l15];
    float bc1 = bias[col0 + 16 + l15];
    store_bf8(dstb, N_, row0,      col0,      lane, t.d00, 1.f, bc0);
    store_bf8(dstb, N_, row0,      col0 + 16, lane, t.d01, 1.f, bc1);
    store_bf8(dstb, N_, row0 + 16, col0,      lane, t.d10, 1.f, bc0);
    store_bf8(dstb, N_, row0 + 16, col0 + 16, lane, t.d11, 1.f, bc1);
  }
}

// ---- stage 4: Mt[b][f][e] = (sum_n k[b][e][n] * v[b][f][n]) / N -----------
__global__ void kv_kernel(const __bf16* __restrict__ k, const __bf16* __restrict__ v,
                          __bf16* __restrict__ mt) {
  int wid  = blockIdx.x * 8 + (threadIdx.x >> 5);
  int lane = threadIdx.x & 31;
  int b  = wid / 64;
  int r  = wid % 64;
  int et2 = r / 8, ft2 = r % 8;                       // rows=e, cols=f
  int row0 = et2 * 32, col0 = ft2 * 32;
  const __bf16* kb = k + (size_t)b * E_ * N_;
  const __bf16* vb = v + (size_t)b * E_ * N_;
  Acc4 t = gemm_tile_2x2(kb, N_, row0, vb, N_, col0, N_, lane);
  __bf16* mb = mt + (size_t)b * E_ * E_;
  const float sc = 1.0f / (float)N_;
  store_bf8(mb, E_, row0,      col0,      lane, t.d00, sc, 0.f);
  store_bf8(mb, E_, row0,      col0 + 16, lane, t.d01, sc, 0.f);
  store_bf8(mb, E_, row0 + 16, col0,      lane, t.d10, sc, 0.f);
  store_bf8(mb, E_, row0 + 16, col0 + 16, lane, t.d11, sc, 0.f);
}

// ---- stage 5: embed[b][n][f] = sum_e q[b][n][e] * M[e][f] -----------------
// rows=f (A = Mt), cols=n (B = q)
__global__ void embed_kernel(const __bf16* __restrict__ mt, const __bf16* __restrict__ q,
                             __bf16* __restrict__ emb) {
  int wid  = blockIdx.x * 8 + (threadIdx.x >> 5);
  int lane = threadIdx.x & 31;
  int b  = wid / 784;
  int r  = wid % 784;
  int ft2 = r / 98, nt2 = r % 98;
  int row0 = ft2 * 32, col0 = nt2 * 32;
  Acc4 t = gemm_tile_2x2(mt + (size_t)b * E_ * E_, E_, row0,
                         q  + (size_t)b * N_ * E_, E_, col0, E_, lane);
  __bf16* eb = emb + (size_t)b * N_ * E_;
  store_bf8(eb, E_, row0,      col0,      lane, t.d00, 1.f, 0.f);
  store_bf8(eb, E_, row0,      col0 + 16, lane, t.d01, 1.f, 0.f);
  store_bf8(eb, E_, row0 + 16, col0,      lane, t.d10, 1.f, 0.f);
  store_bf8(eb, E_, row0 + 16, col0 + 16, lane, t.d11, 1.f, 0.f);
}

// f32 store of one 16x16 tile with bias + residual, rows contiguous.
__device__ __forceinline__ void store_out(float* __restrict__ out,
                                          const float* __restrict__ x,
                                          const float* __restrict__ bo,
                                          size_t bOff, int row0, int col0,
                                          int lane, const v8f acc) {
  int hi8 = ((lane >> 4) & 1) * 8, l15 = lane & 15;
  int c = col0 + l15;
  float bias = bo[c];
  size_t off = bOff + (size_t)c * N_ + row0 + hi8;
  v4f x0 = *(const v4f*)(x + off);
  v4f x1 = *(const v4f*)(x + off + 4);
  v4f o0, o1;
  #pragma unroll
  for (int i = 0; i < 4; ++i) { o0[i] = acc[i] + bias + x0[i];
                                o1[i] = acc[i + 4] + bias + x1[i]; }
  *(v4f*)(out + off)     = o0;
  *(v4f*)(out + off + 4) = o1;
}

// ---- stage 6: out[b][c][n] = sum_f wo[c][f] embed[b][n][f] + bo[c] + x ----
// rows=n (A = embed), cols=c (B = wo)
__global__ void out_kernel(const __bf16* __restrict__ emb, const __bf16* __restrict__ wo,
                           const float* __restrict__ bo, const float* __restrict__ x,
                           float* __restrict__ out) {
  int wid  = blockIdx.x * 8 + (threadIdx.x >> 5);
  int lane = threadIdx.x & 31;
  int b  = wid / 1568;                                // 98 * 16 blocks per batch
  int r  = wid % 1568;
  int nt2 = r / 16, ct2 = r % 16;
  int row0 = nt2 * 32, col0 = ct2 * 32;
  Acc4 t = gemm_tile_2x2(emb + (size_t)b * N_ * E_, E_, row0,
                         wo, E_, col0, E_, lane);
  size_t bOff = (size_t)b * C_ * N_;
  store_out(out, x, bo, bOff, row0,      col0,      lane, t.d00);
  store_out(out, x, bo, bOff, row0,      col0 + 16, lane, t.d01);
  store_out(out, x, bo, bOff, row0 + 16, col0,      lane, t.d10);
  store_out(out, x, bo, bOff, row0 + 16, col0 + 16, lane, t.d11);
}

extern "C" void kernel_launch(void* const* d_in, const int* in_sizes, int n_in,
                              void* d_out, int out_size, void* d_ws, size_t ws_size,
                              hipStream_t stream) {
  const float* x  = (const float*)d_in[0];
  const float* wq = (const float*)d_in[1];
  const float* bq = (const float*)d_in[2];
  const float* wk = (const float*)d_in[3];
  const float* bk = (const float*)d_in[4];
  const float* wv = (const float*)d_in[5];
  const float* bv = (const float*)d_in[6];
  const float* wo = (const float*)d_in[7];
  const float* bo = (const float*)d_in[8];
  float* out = (float*)d_out;

  char* ws = (char*)d_ws;
  __bf16* xt  = (__bf16*)(ws + OFF_XT);
  __bf16* q   = (__bf16*)(ws + OFF_Q);
  __bf16* k   = (__bf16*)(ws + OFF_K);
  __bf16* v   = (__bf16*)(ws + OFF_V);
  __bf16* mt  = (__bf16*)(ws + OFF_MT);
  __bf16* emb = (__bf16*)(ws + OFF_EMB);
  __bf16* wbf = (__bf16*)(ws + OFF_W);

  // 1) transpose-convert x -> xt (bf16)
  cvt_x_kernel<<<dim3(N_ / 32, C_ / 32, B_), dim3(32, 8), 0, stream>>>(x, xt);
  // 2) convert weights
  cvt_w_kernel<<<(4 * E_ * C_) / 256, 256, 0, stream>>>(wq, wk, wv, wo, wbf);
  // 3) Q/K/V projections: 8 * 3 * 8 * 98 = 18816 waves (32x32 block each)
  proj_kernel<<<18816 / 8, 256, 0, stream>>>(xt, wbf, bq, bk, bv, q, k, v);
  // 4) Mt = (K V^T)/N : 8 * 8 * 8 = 512 waves
  kv_kernel<<<512 / 8, 256, 0, stream>>>(k, v, mt);
  // 5) embed = Q M : 8 * 8 * 98 = 6272 waves
  embed_kernel<<<6272 / 8, 256, 0, stream>>>(mt, q, emb);
  // 6) out = wo embed^T + bo + x : 8 * 98 * 16 = 12544 waves
  out_kernel<<<12544 / 8, 256, 0, stream>>>(emb, wbf + (size_t)3 * E_ * C_, bo, x, out);
}
